// Pointnet2ClsMSG_67448166416422
// MI455X (gfx1250) — matrix operand order repaired and checked by reading, not compiled, not run on hardware
//
#include <hip/hip_runtime.h>
#include <hip/hip_bf16.h>
#include <stdint.h>

typedef __attribute__((ext_vector_type(16))) _Float16 v16h;
typedef __attribute__((ext_vector_type(8)))  _Float16 v8h;
typedef __attribute__((ext_vector_type(8)))  float    v8f;

// ---------------------------------------------------------------------------
// Per-layer parameters for a 3-layer pointwise MLP (conv1x1 + BN-style affine
// + relu), weights pre-converted to fp16 with K padded to a multiple of 32.
// ---------------------------------------------------------------------------
struct LayerP {
    const _Float16* W;      // (Cout, Kpad) fp16 row-major
    const float*    gamma;  // (Cout)
    const float*    beta;   // (Cout)
    int Kpad;
    int Cout;
};
struct MlpP { LayerP L[3]; };

// ---------------------------------------------------------------------------
// Weight conversion: f32 (Cout,Kin) -> fp16 (Cout,Kpad), zero padded.
// ---------------------------------------------------------------------------
__global__ __launch_bounds__(256)
void convw_kernel(const float* __restrict__ W, _Float16* __restrict__ O,
                  int Cout, int Kin, int Kpad)
{
    int t = blockIdx.x * 256 + threadIdx.x;
    if (t >= Cout * Kpad) return;
    int r = t / Kpad;
    int k = t - r * Kpad;
    O[t] = (k < Kin) ? (_Float16)W[(size_t)r * Kin + k] : (_Float16)0.0f;
}

// ---------------------------------------------------------------------------
// Farthest point sampling. One block per batch. dist[] lives in LDS; argmax
// tree reduction with lowest-index tie-break (matches jnp.argmax).
// Emits new_xyz directly (fps indices are only needed to gather centers).
// ---------------------------------------------------------------------------
__global__ __launch_bounds__(256)
void fps_kernel(const float* __restrict__ xyz, int stride, int n,
                int npoint, float* __restrict__ newxyz)
{
    const int b = blockIdx.x;
    const int tid = threadIdx.x;
    const int T = 256;
    extern __shared__ unsigned char smraw[];
    float* dist = (float*)smraw;      // n floats
    float* rv   = dist + n;           // T floats
    int*   ri   = (int*)(rv + T);     // T ints

    for (int i = tid; i < n; i += T) dist[i] = 1e10f;
    __syncthreads();

    int far = 0;
    for (int it = 0; it < npoint; ++it) {
        const float* fp = xyz + ((size_t)b * n + far) * stride;
        float fx = fp[0], fy = fp[1], fz = fp[2];
        if (tid == 0) {
            float* o = newxyz + ((size_t)b * npoint + it) * 3;
            o[0] = fx; o[1] = fy; o[2] = fz;
        }
        float bv = -1.0f; int bi = n;
        for (int i = tid; i < n; i += T) {
            const float* p = xyz + ((size_t)b * n + i) * stride;
            float dx = p[0] - fx, dy = p[1] - fy, dz = p[2] - fz;
            float d  = dx * dx + dy * dy + dz * dz;
            float nd = fminf(dist[i], d);
            dist[i] = nd;
            if (nd > bv) { bv = nd; bi = i; }   // strided i ascending -> first max per thread
        }
        rv[tid] = bv; ri[tid] = bi;
        __syncthreads();
        for (int s = T / 2; s > 0; s >>= 1) {
            if (tid < s) {
                float ov = rv[tid + s]; int oi = ri[tid + s];
                if (ov > rv[tid] || (ov == rv[tid] && oi < ri[tid])) {
                    rv[tid] = ov; ri[tid] = oi;
                }
            }
            __syncthreads();
        }
        far = ri[0];
        __syncthreads();
    }
}

// ---------------------------------------------------------------------------
// Ball query. One wave32 per (b, center). Scans points in chunks of 32; keeps
// the first `ns` indices (ascending) inside the radius; fills the remainder
// with the first found index (or 0 if none) -- matches the reference's
// sort+first-fill semantics.
// ---------------------------------------------------------------------------
__global__ __launch_bounds__(128)
void ballq_kernel(const float* __restrict__ xyz, int stride, int n,
                  const float* __restrict__ centers, int ngpb, int total,
                  float r2, int ns, int* __restrict__ idxout)
{
    const int wid  = blockIdx.x * (blockDim.x >> 5) + (threadIdx.x >> 5);
    const int lane = threadIdx.x & 31;
    if (wid >= total) return;

    const int b = wid / ngpb;
    float cx = centers[(size_t)wid * 3 + 0];
    float cy = centers[(size_t)wid * 3 + 1];
    float cz = centers[(size_t)wid * 3 + 2];
    int* out = idxout + (size_t)wid * ns;

    int cnt = 0, first = -1;
    for (int base = 0; base < n && cnt < ns; base += 32) {
        int p = base + lane;
        bool valid = false;
        if (p < n) {
            const float* q = xyz + ((size_t)b * n + p) * stride;
            float dx = q[0] - cx, dy = q[1] - cy, dz = q[2] - cz;
            valid = (dx * dx + dy * dy + dz * dz) < r2;
        }
        unsigned mm = (unsigned)__ballot(valid);
        int pre = __popc(mm & ((1u << lane) - 1u));
        int pos = cnt + pre;
        if (valid && pos < ns) out[pos] = p;
        if (first < 0 && mm != 0u) first = base + (__ffs(mm) - 1);
        cnt += __popc(mm);
    }
    int start = cnt < ns ? cnt : ns;
    int fill  = first >= 0 ? first : 0;
    for (int j = start + lane; j < ns; j += 32) out[j] = fill;
}

// ---------------------------------------------------------------------------
// Fused gather + 3-layer pointwise MLP + max-pool over the group.
// One wave32 per (batch, center) group. Activations staged in LDS as fp16;
// WMMA f32_16x16x32_f16 does all matmuls; pooled output written once.
//
// A-fragment (16x32 f16, ISA 7.12.2): lanes 0-15 row M=lane, halves [0..7]=K
// kb..kb+7 and [8..15]=K kb+16..kb+23; lanes 16-31 row M=lane-16, K offsets
// +8 / +24.  B-fragment (32x16): lane%16 = column N, lanes<16 hold K kb..kb+7 and kb+16..kb+23,
// lanes>=16 hold K kb+8..kb+15 and kb+24..kb+31 (two 8-half runs of weight row N, same split as A).
// ---------------------------------------------------------------------------
__global__ __launch_bounds__(32)
void sa_fused_kernel(const float* __restrict__ xyz, int xstride, int npts,
                     const float* __restrict__ feats, int fstride, int fC,
                     const float* __restrict__ centers,
                     const int* __restrict__ idx,
                     int ns, int ngpb, int identity,
                     MlpP mlp, int K0, int K0pad, int bufElems,
                     float* __restrict__ outp, int ostride, int ooff)
{
    extern __shared__ unsigned char smraw[];
    const int lane = threadIdx.x;
    const int grp  = blockIdx.x;
    const int b    = grp / ngpb;
    const int CoutLast = mlp.L[2].Cout;

    float*     pool = (float*)smraw;
    _Float16*  buf0 = (_Float16*)(pool + CoutLast);
    _Float16*  buf1 = buf0 + bufElems;

    for (int c = lane; c < CoutLast; c += 32) pool[c] = -1e30f;

    float cx = 0.f, cy = 0.f, cz = 0.f;
    if (!identity) {
        cx = centers[(size_t)grp * 3 + 0];
        cy = centers[(size_t)grp * 3 + 1];
        cz = centers[(size_t)grp * 3 + 2];
    }
    const int* gidx = identity ? nullptr : (idx + (size_t)grp * ns);

    const int hi = lane >> 4;   // 0 or 1
    const int lo = lane & 15;

    const int nrt = ns >> 4;    // 16-row tiles
    for (int rt = 0; rt < nrt; ++rt) {
        __syncthreads();
        // ---- gather this row tile into buf0 (16 x K0pad fp16, zero padded)
        for (int t = lane; t < 16 * K0pad; t += 32) {
            int r = t / K0pad, col = t - r * K0pad;
            float v = 0.f;
            if (col < K0) {
                int j = rt * 16 + r;
                int p = identity ? j : gidx[j];
                if (col < 3) {
                    v = xyz[((size_t)b * npts + p) * xstride + col];
                    if (!identity) v -= (col == 0 ? cx : (col == 1 ? cy : cz));
                } else {
                    v = feats[((size_t)b * npts + p) * fstride + (col - 3)];
                }
            }
            buf0[r * K0pad + col] = (_Float16)v;
        }
        __syncthreads();

        const _Float16* src = buf0;
        _Float16* dst = buf1;
        int Kin = K0pad;
        #pragma unroll
        for (int l = 0; l < 3; ++l) {
            const LayerP& lp = mlp.L[l];
            const bool last = (l == 2);
            const int ntiles = lp.Cout >> 4;
            const _Float16* ap = src + lo * Kin + (hi ? 8 : 0);
            for (int nt = 0; nt < ntiles; ++nt) {
                v8f acc = {};
                const _Float16* bp = lp.W + (size_t)(nt * 16 + lo) * Kin + (hi ? 8 : 0);
                for (int kb = 0; kb < Kin; kb += 32) {
                    v8h a0 = *(const v8h*)(ap + kb);
                    v8h a1 = *(const v8h*)(ap + kb + 16);
                    v8h b0 = *(const v8h*)(bp + kb);
                    v8h b1 = *(const v8h*)(bp + kb + 16);
                    v16h A, Bm;
                    #pragma unroll
                    for (int i = 0; i < 8; ++i) {
                        A[i] = a0[i];  A[i + 8] = a1[i];
                        Bm[i] = b0[i]; Bm[i + 8] = b1[i];
                    }
                    acc = __builtin_amdgcn_wmma_f32_16x16x32_f16(
                        false, A, false, Bm, (short)0, acc, false, false);
                }
                const int c = nt * 16 + lo;
                const float g  = lp.gamma[c];
                const float be = lp.beta[c];
                if (!last) {
                    #pragma unroll
                    for (int j = 0; j < 8; ++j) {
                        float v = fmaxf(fmaf(acc[j], g, be), 0.f);
                        dst[(j + hi * 8) * lp.Cout + c] = (_Float16)v;
                    }
                } else {
                    float mx = -1e30f;
                    #pragma unroll
                    for (int j = 0; j < 8; ++j) {
                        float v = fmaxf(fmaf(acc[j], g, be), 0.f);
                        mx = fmaxf(mx, v);
                    }
                    mx = fmaxf(mx, __shfl_xor(mx, 16));
                    if (lane < 16) pool[c] = fmaxf(pool[c], mx);  // single writer
                }
            }
            __syncthreads();
            src = dst;
            Kin = lp.Cout;
            dst = (dst == buf1) ? buf0 : buf1;
        }
    }
    __syncthreads();
    for (int c = lane; c < CoutLast; c += 32)
        outp[(size_t)grp * ostride + ooff + c] = pool[c];
}

// ---------------------------------------------------------------------------
// Host orchestration
// ---------------------------------------------------------------------------
static inline int imax3(int a, int b, int c) {
    int m = a > b ? a : b; return m > c ? m : c;
}

extern "C" void kernel_launch(void* const* d_in, const int* in_sizes, int n_in,
                              void* d_out, int out_size, void* d_ws, size_t ws_size,
                              hipStream_t stream)
{
    (void)in_sizes; (void)n_in; (void)out_size; (void)ws_size;
    const float* pc = (const float*)d_in[0];   // (32, 4096, 9)

    // Layer table: stage1 (3 mlps), stage2 (3 mlps), global; flatten order of
    // the params pytree is (W,gamma,beta) per layer, depth-first.
    static const int KIN[21]  = {   9, 32, 32,    9, 64, 64,    9, 64, 96,
                                  323, 64, 64,  323,128,128,  323,128,128,
                                  643,256,512 };
    static const int COUT[21] = {  32, 32, 64,   64, 64,128,   64, 96,128,
                                   64, 64,128,  128,128,256,  128,128,256,
                                  256,512,1024 };

    auto align256 = [](size_t x) { return (x + 255) & ~(size_t)255; };
    uint8_t* ws = (uint8_t*)d_ws;
    size_t off = 0;
    _Float16* w16[21];
    int kpad[21];
    for (int l = 0; l < 21; ++l) {
        kpad[l] = (KIN[l] + 31) & ~31;
        w16[l] = (_Float16*)(ws + off);
        off = align256(off + (size_t)COUT[l] * kpad[l] * sizeof(_Float16));
    }
    float* newxyz1 = (float*)(ws + off); off = align256(off + (size_t)32*512*3  *4);
    float* feats1  = (float*)(ws + off); off = align256(off + (size_t)32*512*320*4);
    float* newxyz2 = (float*)(ws + off); off = align256(off + (size_t)32*128*3  *4);
    float* feats2  = (float*)(ws + off); off = align256(off + (size_t)32*128*640*4);
    int*   idxbuf  = (int*)  (ws + off); off = align256(off + (size_t)32*512*128*4);

    // 1) convert all weights to fp16 (K padded)
    for (int l = 0; l < 21; ++l) {
        const float* W = (const float*)d_in[1 + l * 3];
        int tot = COUT[l] * kpad[l];
        convw_kernel<<<(tot + 255) / 256, 256, 0, stream>>>(W, w16[l], COUT[l], KIN[l], kpad[l]);
    }

    auto makeMlp = [&](int base) {
        MlpP m;
        for (int i = 0; i < 3; ++i) {
            int l = base + i;
            m.L[i].W     = w16[l];
            m.L[i].gamma = (const float*)d_in[1 + l * 3 + 1];
            m.L[i].beta  = (const float*)d_in[1 + l * 3 + 2];
            m.L[i].Kpad  = kpad[l];
            m.L[i].Cout  = COUT[l];
        }
        return m;
    };

    // 2) stage 1: FPS over 4096 -> 512 centers
    fps_kernel<<<32, 256, (size_t)4096*4 + 256*8, stream>>>(pc, 9, 4096, 512, newxyz1);

    const float R1[3]   = {0.1f, 0.2f, 0.4f};
    const int   NS1[3]  = {16, 32, 128};
    const int   OFF1[3] = {0, 64, 192};
    for (int r = 0; r < 3; ++r) {
        int total = 32 * 512;
        ballq_kernel<<<(total + 3) / 4, 128, 0, stream>>>(
            pc, 9, 4096, newxyz1, 512, total, R1[r] * R1[r], NS1[r], idxbuf);
        MlpP m = makeMlp(r * 3);
        int K0 = 9, K0p = 32;
        int bufElems = 16 * imax3(K0p, m.L[0].Cout, m.L[1].Cout);
        size_t sm = (size_t)m.L[2].Cout * 4 + (size_t)bufElems * 2 * sizeof(_Float16);
        sa_fused_kernel<<<total, 32, sm, stream>>>(
            pc, 9, 4096, pc + 3, 9, 6,
            newxyz1, idxbuf, NS1[r], 512, 0,
            m, K0, K0p, bufElems, feats1, 320, OFF1[r]);
    }

    // 3) stage 2: FPS over 512 -> 128 centers
    fps_kernel<<<32, 256, (size_t)512*4 + 256*8, stream>>>(newxyz1, 3, 512, 128, newxyz2);

    const float R2[3]   = {0.2f, 0.4f, 0.8f};
    const int   NS2[3]  = {32, 64, 128};
    const int   OFF2[3] = {0, 128, 384};
    for (int r = 0; r < 3; ++r) {
        int total = 32 * 128;
        ballq_kernel<<<(total + 3) / 4, 128, 0, stream>>>(
            newxyz1, 3, 512, newxyz2, 128, total, R2[r] * R2[r], NS2[r], idxbuf);
        MlpP m = makeMlp(9 + r * 3);
        int K0 = 323, K0p = 352;
        int bufElems = 16 * imax3(K0p, m.L[0].Cout, m.L[1].Cout);
        size_t sm = (size_t)m.L[2].Cout * 4 + (size_t)bufElems * 2 * sizeof(_Float16);
        sa_fused_kernel<<<total, 32, sm, stream>>>(
            newxyz1, 3, 512, feats1, 320, 320,
            newxyz2, idxbuf, NS2[r], 128, 0,
            m, K0, K0p, bufElems, feats2, 640, OFF2[r]);
    }

    // 4) global MLP over the 128 stage-2 points per batch (identity grouping),
    //    pooled result (32 x 1024) goes straight to d_out.
    {
        MlpP m = makeMlp(18);
        int K0 = 643, K0p = 672;
        int bufElems = 16 * imax3(K0p, m.L[0].Cout, m.L[1].Cout);
        size_t sm = (size_t)m.L[2].Cout * 4 + (size_t)bufElems * 2 * sizeof(_Float16);
        sa_fused_kernel<<<32, 32, sm, stream>>>(
            newxyz2, 3, 128, feats2, 640, 640,
            nullptr, nullptr, 128, 1, 1,
            m, K0, K0p, bufElems, (float*)d_out, 1024, 0);
    }
}
